// ParityACTModel_37117107372578
// MI455X (gfx1250) — compile-verified, matrix operand order checked by reading, not run on hardware
//
#include <hip/hip_runtime.h>
#include <hip/hip_bf16.h>

typedef __bf16 bf16_t;
typedef __attribute__((ext_vector_type(16))) bf16_t v16bf;
typedef __attribute__((ext_vector_type(8)))  float  v8f;
typedef __attribute__((ext_vector_type(4)))  unsigned int u32x4;
typedef __attribute__((ext_vector_type(8)))  int i32x8;
typedef __attribute__((ext_vector_type(4)))  int i32x4;

union FragBF {
    v16bf v;
    uint4 u[2];
};

#define B_SZ 4096
#define D_SZ 512
#define H_SZ 1024
#define MAXP 20

#define M_ROWS 64              // batch rows per block
#define BLK    512             // threads per block (16 waves)

#define PANEL_ELEMS (H_SZ * 32)

// staging tier: 2 = TDM, 1 = async global->LDS, 0 = synchronous copy
#if __has_builtin(__builtin_amdgcn_tensor_load_to_lds)
#define STAGE_MODE 2
#warning "CDNA5 staging: STAGE_MODE=2 (tensor_load_to_lds / TDM)"
#elif __has_builtin(__builtin_amdgcn_global_load_async_to_lds_b128)
#define STAGE_MODE 1
#warning "CDNA5 staging: STAGE_MODE=1 (global_load_async_to_lds)"
#else
#define STAGE_MODE 0
#warning "CDNA5 staging: STAGE_MODE=0 (synchronous copy fallback)"
#endif

// workspace layout (bytes)
#define WS_WHH   ((size_t)0)                        // 2 MB  bf16 [H][H]
#define WS_WIH   ((size_t)2 << 20)                  // 1 MB  bf16 [H][D]
#define WS_XBF   ((size_t)3 << 20)                  // 4 MB  bf16 [B][D]
#define WS_WFLAG ((size_t)7 << 20)                  // 4 KB  f32  [H]
#define WS_XIHB  (((size_t)7 << 20) + 4096)         // 16 MB f32  [B][H]
#define WS_ACC   (WS_XIHB + ((size_t)16 << 20))     // 16 MB f32  [B][H]

// ---------------- prologue: conversions ----------------

__global__ void cvt_f32_bf16(const float* __restrict__ s, bf16_t* __restrict__ d, int n) {
    int i = blockIdx.x * 256 + threadIdx.x;
    if (i < n) d[i] = (bf16_t)s[i];
}

__global__ void pack_wih(const float* __restrict__ wih, bf16_t* __restrict__ wihbf,
                         float* __restrict__ wflag) {
    int i = blockIdx.x * 256 + threadIdx.x;
    if (i < H_SZ * D_SZ) {
        int r = i >> 9;           // D_SZ == 512
        int c = i & (D_SZ - 1);
        wihbf[i] = (bf16_t)wih[r * (D_SZ + 1) + c];
    }
    if (i < H_SZ) wflag[i] = wih[i * (D_SZ + 1) + D_SZ];
}

// ---------------- x_ih = x @ W_ih[:, :D].T + b_ih + b_hh ----------------

__global__ void __launch_bounds__(256)
xih_gemm(const bf16_t* __restrict__ xbf, const bf16_t* __restrict__ wihbf,
         const float* __restrict__ bih, const float* __restrict__ bhh,
         float* __restrict__ xihb) {
    const int w = threadIdx.x >> 5, lane = threadIdx.x & 31;
    const int lr = lane & 15, half = lane >> 4;
    const int tidg = blockIdx.x * 8 + w;
    const int mtile = tidg >> 6;   // 0..255
    const int ntile = tidg & 63;   // 0..63

    const bf16_t* arow = xbf   + (size_t)(mtile * 16 + lr) * D_SZ;
    const bf16_t* brow = wihbf + (size_t)(ntile * 16 + lr) * D_SZ;

    v8f acc = (v8f){0.f,0.f,0.f,0.f,0.f,0.f,0.f,0.f};
    for (int kc = 0; kc < D_SZ / 32; ++kc) {
        FragBF a, b;
        int ko = kc * 32 + (half ? 8 : 0);
        a.u[0] = *(const uint4*)(arow + ko);
        a.u[1] = *(const uint4*)(arow + ko + 16);
        b.u[0] = *(const uint4*)(brow + ko);
        b.u[1] = *(const uint4*)(brow + ko + 16);
        acc = __builtin_amdgcn_wmma_f32_16x16x32_bf16(false, a.v, false, b.v,
                                                      (short)0, acc, false, false);
    }
    const int col = ntile * 16 + lr;
    const float bsum = bih[col] + bhh[col];
    for (int i = 0; i < 8; ++i) {
        int row = mtile * 16 + half * 8 + i;
        xihb[(size_t)row * H_SZ + col] = acc[i] + bsum;
    }
}

// ---------------- staging helpers ----------------

#if STAGE_MODE == 2
// Tensor Data Mover: load W_hh[:, kc*32 .. +32) (1024 rows x 64B) into LDS.
// D# built per CDNA5 ISA 8.3/8.4: 2D tile, data_size=2B, groups 2/3 zero.
__device__ __forceinline__ void tdm_load_panel(const bf16_t* whhbf, int kc,
                                               unsigned lds_byte) {
    unsigned long long ga = (unsigned long long)(uintptr_t)(whhbf + kc * 32);
    u32x4 g0;
    g0[0] = 1u;                                            // count=1 (user mode)
    g0[1] = lds_byte;                                      // lds_addr
    g0[2] = (unsigned)ga;                                  // global_addr[31:0]
    g0[3] = ((unsigned)(ga >> 32) & 0x01FFFFFFu)           // global_addr[56:32]
            | 0x80000000u;                                 // type=2 ("image")
    i32x8 g1;
    g1[0] = 0x00010000;                                    // data_size=1 (2 bytes)
    g1[1] = (int)((unsigned)H_SZ << 16);                   // tensor_dim0 = 1024 (k)
    g1[2] = (int)((unsigned)H_SZ << 16);                   // tensor_dim1 = 1024 (n)
    g1[3] = (int)(32u << 16);                              // tile_dim0 = 32
    g1[4] = H_SZ;                                          // tile_dim1 = 1024, tile_dim2 = 0
    g1[5] = H_SZ;                                          // tensor_dim0_stride = 1024
    g1[6] = 0;
    g1[7] = 0;
    i32x4 z4 = (i32x4){0, 0, 0, 0};
#if __clang_major__ >= 23
    i32x8 z8 = (i32x8){0, 0, 0, 0, 0, 0, 0, 0};
    __builtin_amdgcn_tensor_load_to_lds(g0, g1, z4, z4, z8, 0);
#else
    __builtin_amdgcn_tensor_load_to_lds(g0, g1, z4, z4, 0);
#endif
}
#endif

#if STAGE_MODE == 1
__device__ __forceinline__ void async_stage_panel(const bf16_t* whhbf, int kc,
                                                  bf16_t* dst, int tid) {
    typedef __attribute__((address_space(1))) const void gvoid_t;
    typedef __attribute__((address_space(3))) void lvoid_t;
    for (int r = 0; r < 2; ++r) {
        int n = tid * 2 + r;
        const bf16_t* src = whhbf + (size_t)n * H_SZ + kc * 32;
        bf16_t* d = dst + n * 32;
        for (int c = 0; c < 4; ++c) {
            __builtin_amdgcn_global_load_async_to_lds_b128(
                (gvoid_t*)(uintptr_t)(src + c * 8),
                (lvoid_t*)(uintptr_t)(unsigned)(uintptr_t)(d + c * 8),
                0, 0);
        }
    }
}
__device__ __forceinline__ void wait_async(int n) {
#if __has_builtin(__builtin_amdgcn_s_wait_asynccnt)
    if (n == 0) __builtin_amdgcn_s_wait_asynccnt(0);
    else        __builtin_amdgcn_s_wait_asynccnt(8);
#else
    if (n == 0) asm volatile("s_wait_asynccnt 0" ::: "memory");
    else        asm volatile("s_wait_asynccnt 8" ::: "memory");
#endif
}
#endif

// ---------------- main ACT ponder loop ----------------
// 64 blocks x 512 threads (16 waves); each block owns 64 batch rows.

__global__ void __launch_bounds__(BLK)
act_main(const bf16_t* __restrict__ whhbf, const float* __restrict__ xihb,
         const float* __restrict__ wflag, float* __restrict__ accg,
         const float* __restrict__ wp, const float* __restrict__ fc1w,
         const float* __restrict__ bpp, const float* __restrict__ fc1b,
         float* __restrict__ dout) {
    __shared__ __align__(16) bf16_t sh_hx[M_ROWS * H_SZ];      // 128 KB
    __shared__ __align__(16) bf16_t sh_panel[2][PANEL_ELEMS];  // 2 x 64 KB double buffer
    __shared__ float sh_wp[H_SZ];
    __shared__ float sh_fc1[H_SZ];
    __shared__ float sh_active[M_ROWS], sh_coef[M_ROWS], sh_ah[M_ROWS],
                     sh_spc[M_ROWS], sh_sc[M_ROWS];
    __shared__ int sh_any;

    const int tid = threadIdx.x;
    const int w = tid >> 5, lane = tid & 31;
    const int lr = lane & 15, half = lane >> 4;
    const int mt = w & 3;      // which 16-row M tile (0..3)
    const int ng = w >> 2;     // which 256-column N group (0..3)
    const int row0 = blockIdx.x * M_ROWS;

    for (int i = tid; i < M_ROWS * H_SZ; i += BLK) sh_hx[i] = (bf16_t)0.0f;
    for (int i = tid; i < H_SZ; i += BLK) { sh_wp[i] = wp[i]; sh_fc1[i] = fc1w[i]; }
    for (int i = tid; i < M_ROWS * H_SZ; i += BLK) accg[(size_t)row0 * H_SZ + i] = 0.0f;
    if (tid < M_ROWS) {
        sh_active[tid] = 1.f; sh_coef[tid] = 0.f; sh_ah[tid] = 0.f;
        sh_spc[tid] = 0.f;    sh_sc[tid] = 0.f;
    }
    __syncthreads();

    int   ncol[16];
    float wf[16];
    for (int j = 0; j < 16; ++j) {
        ncol[j] = ng * 256 + j * 16 + lr;
        wf[j]   = wflag[ncol[j]];
    }
    const float bpv = bpp[0];

    for (int t = 0; t < MAXP; ++t) {
        v8f acc[16];
        for (int j = 0; j < 16; ++j)
            acc[j] = (v8f){0.f,0.f,0.f,0.f,0.f,0.f,0.f,0.f};

#if STAGE_MODE == 2
        if (w == 0) tdm_load_panel(whhbf, 0, (unsigned)(uintptr_t)&sh_panel[0][0]);
#elif STAGE_MODE == 1
        async_stage_panel(whhbf, 0, &sh_panel[0][0], tid);
#endif

        for (int kc = 0; kc < H_SZ / 32; ++kc) {
            const bf16_t* panel = &sh_panel[kc & 1][0];

#if STAGE_MODE == 2
            if (kc + 1 < H_SZ / 32) {
                if (w == 0)
                    tdm_load_panel(whhbf, kc + 1,
                                   (unsigned)(uintptr_t)&sh_panel[(kc + 1) & 1][0]);
                __builtin_amdgcn_s_wait_tensorcnt(1);   // panel kc complete
            } else {
                __builtin_amdgcn_s_wait_tensorcnt(0);
            }
#elif STAGE_MODE == 1
            if (kc + 1 < H_SZ / 32) {
                async_stage_panel(whhbf, kc + 1, &sh_panel[(kc + 1) & 1][0], tid);
                wait_async(8);                          // panel kc complete
            } else {
                wait_async(0);
            }
#else
            for (int r = 0; r < 2; ++r) {
                int n = tid * 2 + r;
                const uint4* s = (const uint4*)(whhbf + (size_t)n * H_SZ + kc * 32);
                uint4* d = (uint4*)((bf16_t*)&sh_panel[kc & 1][0] + n * 32);
                d[0] = s[0]; d[1] = s[1]; d[2] = s[2]; d[3] = s[3];
            }
#endif
            __syncthreads();

            FragBF a;
            {
                const bf16_t* ap = sh_hx + (mt * 16 + lr) * H_SZ + kc * 32 + (half ? 8 : 0);
                a.u[0] = *(const uint4*)ap;
                a.u[1] = *(const uint4*)(ap + 16);
            }
            for (int j = 0; j < 16; ++j) {
                FragBF b;
                const bf16_t* bp2 = panel + ncol[j] * 32 + (half ? 8 : 0);
                b.u[0] = *(const uint4*)bp2;
                b.u[1] = *(const uint4*)(bp2 + 16);
                acc[j] = __builtin_amdgcn_wmma_f32_16x16x32_bf16(false, a.v, false, b.v,
                                                                 (short)0, acc[j], false, false);
            }
            __syncthreads();
        }

        // epilogue: cand = tanh(raw + x_ih + t*w_flag); hx updated where active
        float act8[8];
        for (int i = 0; i < 8; ++i) act8[i] = sh_active[mt * 16 + half * 8 + i];
        const float tstep = (float)t;
        for (int j = 0; j < 16; ++j) {
            const int n = ncol[j];
            for (int i = 0; i < 8; ++i) {
                if (act8[i] != 0.f) {
                    int m = mt * 16 + half * 8 + i;
                    float cand = tanhf(acc[j][i] + xihb[(size_t)(row0 + m) * H_SZ + n]
                                       + tstep * wf[j]);
                    sh_hx[m * H_SZ + n] = (bf16_t)cand;
                }
            }
        }
        __syncthreads();

        // per-row halting: h = sigmoid(hx . w_p + b_p); each wave handles 4 rows
        for (int q = 0; q < 4; ++q) {
            int r = w * 4 + q;
            float s = 0.f;
            for (int k = lane; k < H_SZ; k += 32)
                s += (float)sh_hx[r * H_SZ + k] * sh_wp[k];
            s += __shfl_xor(s, 16); s += __shfl_xor(s, 8); s += __shfl_xor(s, 4);
            s += __shfl_xor(s, 2);  s += __shfl_xor(s, 1);
            if (lane == 0) {
                float a = sh_active[r];
                float h = 1.f / (1.f + expf(-(s + bpv)));
                float ah_old = sh_ah[r];
                if (a != 0.f) sh_spc[r] = ah_old;          // spc = where(active, accum_h, spc)
                float ah_new = ah_old + a * h;
                float p = h - fmaxf(ah_new - 1.f, 0.f);
                sh_coef[r] = a * (1.f + p);                // accum_hx += a*(1+p)*hx
                sh_ah[r]   = ah_new;
                sh_sc[r]  += a;
                sh_active[r] = (ah_new < 1.f - 0.01f) ? 1.f : 0.f;
            }
        }
        __syncthreads();

        // accum_hx update (global scratch, L2-resident) + active count
        for (int i = tid; i < M_ROWS * H_SZ; i += BLK) {
            int m = i >> 10;
            float c = sh_coef[m];
            if (c != 0.f)
                accg[(size_t)row0 * H_SZ + i] += c * (float)sh_hx[i];
        }
        if (tid == 0) {
            int c = 0;
            for (int r = 0; r < M_ROWS; ++r) c += (sh_active[r] != 0.f);
            sh_any = c;
        }
        __syncthreads();
        if (sh_any == 0) break;   // frozen rows never change again: identical math
    }

    // out[b] = (accum_hx[b,:] . fc1_w) / step_count + fc1_b ; ponder = -spc
    for (int q = 0; q < 4; ++q) {
        int r = w * 4 + q;
        float s = 0.f;
        for (int k = lane; k < H_SZ; k += 32)
            s += accg[(size_t)(row0 + r) * H_SZ + k] * sh_fc1[k];
        s += __shfl_xor(s, 16); s += __shfl_xor(s, 8); s += __shfl_xor(s, 4);
        s += __shfl_xor(s, 2);  s += __shfl_xor(s, 1);
        if (lane == 0) {
            dout[row0 + r]        = s / sh_sc[r] + fc1b[0];
            dout[B_SZ + row0 + r] = -sh_spc[r];
        }
    }
}

// ---------------- launch ----------------

extern "C" void kernel_launch(void* const* d_in, const int* in_sizes, int n_in,
                              void* d_out, int out_size, void* d_ws, size_t ws_size,
                              hipStream_t stream) {
    const float* x    = (const float*)d_in[0];
    const float* Wih  = (const float*)d_in[1];
    const float* bih  = (const float*)d_in[2];
    const float* Whh  = (const float*)d_in[3];
    const float* bhh  = (const float*)d_in[4];
    const float* wpv  = (const float*)d_in[5];
    const float* bpv  = (const float*)d_in[6];
    const float* fc1w = (const float*)d_in[7];
    const float* fc1b = (const float*)d_in[8];
    float* out = (float*)d_out;

    char* ws = (char*)d_ws;
    bf16_t* whhbf = (bf16_t*)(ws + WS_WHH);
    bf16_t* wihbf = (bf16_t*)(ws + WS_WIH);
    bf16_t* xbf   = (bf16_t*)(ws + WS_XBF);
    float*  wflag = (float*)(ws + WS_WFLAG);
    float*  xihb  = (float*)(ws + WS_XIHB);
    float*  accg  = (float*)(ws + WS_ACC);

    cvt_f32_bf16<<<(H_SZ * H_SZ + 255) / 256, 256, 0, stream>>>(Whh, whhbf, H_SZ * H_SZ);
    cvt_f32_bf16<<<(B_SZ * D_SZ + 255) / 256, 256, 0, stream>>>(x, xbf, B_SZ * D_SZ);
    pack_wih<<<(H_SZ * D_SZ + 255) / 256, 256, 0, stream>>>(Wih, wihbf, wflag);
    xih_gemm<<<(B_SZ / 16) * (H_SZ / 16) / 8, 256, 0, stream>>>(xbf, wihbf, bih, bhh, xihb);
    act_main<<<B_SZ / M_ROWS, BLK, 0, stream>>>(whhbf, xihb, wflag, accg, wpv, fc1w,
                                                bpv, fc1b, out);
}
